// MultiQueryAttention_29231547416696
// MI455X (gfx1250) — matrix operand order repaired and checked by reading, not compiled, not run on hardware
//
#include <hip/hip_runtime.h>
#include <math.h>

typedef __bf16 bf16_t;
typedef __attribute__((ext_vector_type(16))) __bf16 v16bf;
typedef __attribute__((ext_vector_type(8)))  __bf16 v8bf;
typedef __attribute__((ext_vector_type(8)))  float  v8f;

#define HIDDEN  2048
#define NH      16
#define HD      128
#define BATCH   2
#define SEQ     2048
#define ROWS    (BATCH * SEQ)          // 4096
#define SCALE_ATTN 0.08838834764831845f  // 1/sqrt(128)

// ---------------- CDNA5 async Global->LDS copy (ASYNCcnt path) ----------------
// VDST is the LDS byte address relative to the wave's LDS base; the low 32 bits
// of a flat pointer to __shared__ memory are exactly that offset.
static __device__ __forceinline__ void async_load_b128(void* lds_dst, const void* gsrc) {
  unsigned l = (unsigned)(unsigned long long)lds_dst;
  asm volatile("global_load_async_to_lds_b128 %0, %1, off"
               :: "v"(l), "v"(gsrc) : "memory");
}
static __device__ __forceinline__ void wait_async0() {
  asm volatile("s_wait_asynccnt 0x0" ::: "memory");
}

// ---------------- WMMA helpers (CDNA5 layouts) ----------------

static __device__ __forceinline__ v8f wmma_bf16(v16bf a, v16bf b, v8f c) {
  // v_wmma_f32_16x16x32_bf16, f32 accumulate
  return __builtin_amdgcn_wmma_f32_16x16x32_bf16(false, a, false, b, (short)0, c,
                                                 false, false);
}

// A operand, 16x32 bf16 (M x K), source tile row-major with leading dim `lda`.
// ISA layout: lane&15 = row; lane<16 holds K 0-7 & 16-23, lane>=16 holds K 8-15 & 24-31.
static __device__ __forceinline__ v16bf load_a_frag(const bf16_t* sm, int lda, int lane) {
  const bf16_t* p = sm + (lane & 15) * lda + ((lane >> 4) << 3);
  v8bf lo = *(const v8bf*)(p);        // K base .. base+7
  v8bf hi = *(const v8bf*)(p + 16);   // K base+16 .. base+23
  v16bf a;
#pragma unroll
  for (int i = 0; i < 8; ++i) { a[i] = lo[i]; a[i + 8] = hi[i]; }
  return a;
}

// B operand, 32x16 bf16 (K x N), source stored TRANSPOSED (N x K) row-major, ld `ldb`.
// ISA layout: lane&15 = column; lane<16 holds K 0-7 & 16-23, lane>=16 holds K 8-15 & 24-31.
static __device__ __forceinline__ v16bf load_bt_frag(const bf16_t* sm, int ldb, int lane) {
  const bf16_t* p = sm + (lane & 15) * ldb + ((lane >> 4) << 3);
  v8bf lo = *(const v8bf*)(p);
  v8bf hi = *(const v8bf*)(p + 16);
  v16bf b;
#pragma unroll
  for (int i = 0; i < 8; ++i) { b[i] = lo[i]; b[i + 8] = hi[i]; }
  return b;
}

// ---------------- conversion kernels ----------------

__global__ void cvt_f32_bf16_kernel(const float* __restrict__ src,
                                    bf16_t* __restrict__ dst, int n) {
  int i = blockIdx.x * blockDim.x + threadIdx.x;
  if (i < n) dst[i] = (bf16_t)src[i];
}

// src: K x N f32 row-major  ->  dst: N x K bf16 row-major (transposed)
__global__ __launch_bounds__(256)
void tcvt_f32_bf16_kernel(const float* __restrict__ src, bf16_t* __restrict__ dst,
                          int K, int N) {
  __shared__ float tile[32][33];
  const int n0 = blockIdx.x * 32;
  const int k0 = blockIdx.y * 32;
  const int tx = threadIdx.x & 31;
  const int ty = threadIdx.x >> 5;  // 0..7
#pragma unroll
  for (int r = ty; r < 32; r += 8)
    tile[r][tx] = src[(size_t)(k0 + r) * N + n0 + tx];
  __syncthreads();
#pragma unroll
  for (int r = ty; r < 32; r += 8)
    dst[(size_t)(n0 + r) * K + k0 + tx] = (bf16_t)tile[tx][r];
}

// -------- tiled bf16 WMMA GEMM: C(f32) = A(bf16, MxK) * BT(bf16, NxK)^T --------
// Double-buffered LDS, async Global->LDS staging (ASYNCcnt), one barrier/iter.

#define GT_M 128
#define GT_N 128
#define GT_K 32
#define LDSA 40
#define LDSB 40

__global__ __launch_bounds__(256)
void gemm_bf16_kernel(const bf16_t* __restrict__ A, const bf16_t* __restrict__ BT,
                      float* __restrict__ C, int M, int N, int K) {
  __shared__ bf16_t sA[2][GT_M * LDSA];
  __shared__ bf16_t sB[2][GT_N * LDSB];

  const int tid  = threadIdx.x;
  const int lane = tid & 31;
  const int wid  = tid >> 5;           // 0..7
  const int m0   = blockIdx.y * GT_M;
  const int n0   = blockIdx.x * GT_N;
  const int wm   = (wid & 3) * 32;     // wave row offset within block tile
  const int wn   = (wid >> 2) * 64;    // wave col offset within block tile

  v8f acc[2][4];
#pragma unroll
  for (int i = 0; i < 2; ++i)
#pragma unroll
    for (int j = 0; j < 4; ++j) acc[i][j] = {};

  // staging job: thread -> (row srow, 16-half segment scol) of the 128x32 tiles
  const int srow = tid >> 1;           // 0..127
  const int scol = (tid & 1) * 16;     // 0 or 16 halves

  const bf16_t* gA = A  + (size_t)(m0 + srow) * K + scol;
  const bf16_t* gB = BT + (size_t)(n0 + srow) * K + scol;
  bf16_t* lA[2] = { &sA[0][srow * LDSA + scol], &sA[1][srow * LDSA + scol] };
  bf16_t* lB[2] = { &sB[0][srow * LDSB + scol], &sB[1][srow * LDSB + scol] };

  // prologue: issue tile 0 into buffer 0
  async_load_b128(lA[0],     gA);
  async_load_b128(lA[0] + 8, gA + 8);
  async_load_b128(lB[0],     gB);
  async_load_b128(lB[0] + 8, gB + 8);

  const int iters = K / GT_K;
  for (int it = 0; it < iters; ++it) {
    wait_async0();        // my async writes for tile `it` have landed
    __syncthreads();      // everyone's writes visible; prev-iter readers done
    if (it + 1 < iters) { // overlap: issue tile it+1 while computing tile it
      const int nb = (it + 1) & 1;
      const bf16_t* ga = gA + (size_t)(it + 1) * GT_K;
      const bf16_t* gb = gB + (size_t)(it + 1) * GT_K;
      async_load_b128(lA[nb],     ga);
      async_load_b128(lA[nb] + 8, ga + 8);
      async_load_b128(lB[nb],     gb);
      async_load_b128(lB[nb] + 8, gb + 8);
    }
    const bf16_t* bufA = sA[it & 1];
    const bf16_t* bufB = sB[it & 1];

    v16bf a0 = load_a_frag(bufA + (wm + 0)  * LDSA, LDSA, lane);
    v16bf a1 = load_a_frag(bufA + (wm + 16) * LDSA, LDSA, lane);
#pragma unroll
    for (int j = 0; j < 4; ++j) {
      v16bf b = load_bt_frag(bufB + (wn + 16 * j) * LDSB, LDSB, lane);
      acc[0][j] = wmma_bf16(a0, b, acc[0][j]);
      acc[1][j] = wmma_bf16(a1, b, acc[1][j]);
    }
  }

  // C layout: element r -> row (r + 8*(lane>=16)), col = lane&15
  const int rhi = (lane >> 4) << 3;
  const int cn  = lane & 15;
#pragma unroll
  for (int i = 0; i < 2; ++i)
#pragma unroll
    for (int j = 0; j < 4; ++j)
#pragma unroll
      for (int r = 0; r < 8; ++r)
        C[(size_t)(m0 + wm + i * 16 + rhi + r) * N + n0 + wn + j * 16 + cn] =
            acc[i][j][r];
}

// ---------------- RoPE (faithful repeat_interleave + half-split mix) ----------------

static __device__ __forceinline__ float rope_mix(float x, float partner, int d, float pos) {
  // theta uses freq index d>>1 ; rotate_half pairs (d, d^64)
  float theta = pos * __powf(10000.0f, -(float)(d & ~1) * (1.0f / 128.0f));
  float s, c;
  __sincosf(theta, &s, &c);
  float rot = (d < 64) ? -partner : partner;
  return c * x + s * rot;
}

__global__ void rope_q_kernel(const float* __restrict__ Qraw,
                              const int* __restrict__ pos_ids,
                              bf16_t* __restrict__ Qb) {
  int idx = blockIdx.x * blockDim.x + threadIdx.x;   // B*L*NH*HD = 8388608
  if (idx >= BATCH * SEQ * NH * HD) return;
  const int d = idx & 127;
  const int h = (idx >> 7) & 15;
  const int l = (idx >> 11) & 2047;
  const int b = idx >> 22;
  const size_t row = (size_t)b * SEQ + l;
  const float x  = Qraw[row * HIDDEN + h * HD + d];
  const float xp = Qraw[row * HIDDEN + h * HD + (d ^ 64)];
  const float o  = rope_mix(x, xp, d, (float)pos_ids[l]);
  Qb[(((size_t)b * NH + h) * SEQ + l) * HD + d] = (bf16_t)o;
}

__global__ void rope_kv_kernel(const float* __restrict__ KVraw,
                               const int* __restrict__ pos_ids,
                               bf16_t* __restrict__ Kb,   // [B, L, 128]
                               bf16_t* __restrict__ Vt) { // [B, 128, L]
  int idx = blockIdx.x * blockDim.x + threadIdx.x;   // B*L*256 = 1048576
  if (idx >= BATCH * SEQ * 256) return;
  const int d2 = idx & 255;
  const int l  = (idx >> 8) & 2047;
  const int b  = idx >> 19;
  const size_t row = (size_t)b * SEQ + l;
  const float x = KVraw[row * 256 + d2];
  if (d2 < 128) {
    const float xp = KVraw[row * 256 + (d2 ^ 64)];
    const float o  = rope_mix(x, xp, d2, (float)pos_ids[l]);
    Kb[row * HD + d2] = (bf16_t)o;
  } else {
    const int d = d2 - 128;
    Vt[((size_t)b * HD + d) * SEQ + l] = (bf16_t)x;
  }
}

// ---------------- flash attention (online softmax, WMMA, async staging) ----------------

#define KB 32      // key positions per strip
#define LDK 136    // sK leading dim (128 + 8 pad, keeps 16B alignment)
#define LDV 40     // sV leading dim (32 + 8)
#define LDP 40     // per-wave P scratch leading dim

__global__ __launch_bounds__(128)
void flash_attn_kernel(const bf16_t* __restrict__ Qb,  // [B, NH, L, 128]
                       const bf16_t* __restrict__ Kb,  // [B, L, 128]
                       const bf16_t* __restrict__ Vt,  // [B, 128, L]
                       bf16_t* __restrict__ AO) {      // [B, L, NH*128]
  __shared__ bf16_t sK[2][KB * LDK];     // key strip, row = kpos, col = d
  __shared__ bf16_t sV[2][HD * LDV];     // row = d, col = kpos (Vt slice)
  __shared__ bf16_t sP[4 * 16 * LDP];    // per-wave C->A transpose scratch

  const int tid  = threadIdx.x;
  const int lane = tid & 31;
  const int wid  = tid >> 5;                 // 0..3
  const int b    = blockIdx.z;
  const int h    = blockIdx.y;
  const int q0   = blockIdx.x * 64 + wid * 16;

  const bf16_t* Qh  = Qb + (((size_t)b * NH + h) * SEQ) * HD;
  const bf16_t* Kbb = Kb + (size_t)b * SEQ * HD;
  const bf16_t* Vtb = Vt + (size_t)b * HD * SEQ;

  // Q fragments for this wave's 16 rows, resident across the whole k loop
  v16bf qf[4];
  {
    const bf16_t* qp = Qh + (size_t)(q0 + (lane & 15)) * HD + ((lane >> 4) << 3);
#pragma unroll
    for (int kc = 0; kc < 4; ++kc) {
      v8bf lo = *(const v8bf*)(qp + kc * 32);
      v8bf hi = *(const v8bf*)(qp + kc * 32 + 16);
#pragma unroll
      for (int i = 0; i < 8; ++i) { qf[kc][i] = lo[i]; qf[kc][i + 8] = hi[i]; }
    }
  }

  v8f o[8];
#pragma unroll
  for (int n = 0; n < 8; ++n) o[n] = {};
  float m[8], lsum[8];
#pragma unroll
  for (int r = 0; r < 8; ++r) { m[r] = -__builtin_inff(); lsum[r] = 0.0f; }

  const int rhi = (lane >> 4) << 3;
  const int cn  = lane & 15;

  // staging jobs: K strip -> thread (tid>>2, (tid&3)*32 halves); V slice -> row tid
  const int krow = tid >> 2;
  const int kseg = (tid & 3) * 32;
  const bf16_t* gK = Kbb + (size_t)krow * HD + kseg;
  const bf16_t* gV = Vtb + (size_t)tid * SEQ;
  bf16_t* lK[2] = { &sK[0][krow * LDK + kseg], &sK[1][krow * LDK + kseg] };
  bf16_t* lV[2] = { &sV[0][tid * LDV],         &sV[1][tid * LDV] };

  // prologue: stage strip 0 into buffer 0 (4x b128 for K, 4x for V per thread)
#pragma unroll
  for (int c = 0; c < 4; ++c) {
    async_load_b128(lK[0] + 8 * c, gK + 8 * c);
    async_load_b128(lV[0] + 8 * c, gV + 8 * c);
  }

  const int iters = SEQ / KB;  // 64
  for (int it = 0; it < iters; ++it) {
    wait_async0();
    __syncthreads();
    if (it + 1 < iters) {
      const int nb = (it + 1) & 1;
      const bf16_t* gk = gK + (size_t)(it + 1) * KB * HD;  // next 32 key rows
      const bf16_t* gv = gV + (size_t)(it + 1) * KB;       // next 32 kpos cols
#pragma unroll
      for (int c = 0; c < 4; ++c) {
        async_load_b128(lK[nb] + 8 * c, gk + 8 * c);
        async_load_b128(lV[nb] + 8 * c, gv + 8 * c);
      }
    }
    const bf16_t* bufK = sK[it & 1];
    const bf16_t* bufV = sV[it & 1];

    // ---- S = Q * K^T for this strip: two 16x16 tiles, K-dim = 128 ----
    v8f s0 = {}, s1 = {};
#pragma unroll
    for (int kc = 0; kc < 4; ++kc) {
      v16bf bk0 = load_bt_frag(bufK + 0  * LDK + kc * 32, LDK, lane);
      v16bf bk1 = load_bt_frag(bufK + 16 * LDK + kc * 32, LDK, lane);
      s0 = wmma_bf16(qf[kc], bk0, s0);
      s1 = wmma_bf16(qf[kc], bk1, s1);
    }

    // ---- online softmax update (rows in elements, cols in 16-lane halves) ----
    float corr[8];
#pragma unroll
    for (int r = 0; r < 8; ++r) {
      float v0 = s0[r] * SCALE_ATTN;
      float v1 = s1[r] * SCALE_ATTN;
      float mt = fmaxf(v0, v1);
#pragma unroll
      for (int off = 1; off < 16; off <<= 1) mt = fmaxf(mt, __shfl_xor(mt, off, 32));
      const float mnew = fmaxf(m[r], mt);
      corr[r] = __expf(m[r] - mnew);
      const float p0 = __expf(v0 - mnew);
      const float p1 = __expf(v1 - mnew);
      s0[r] = p0; s1[r] = p1;
      float rs = p0 + p1;
#pragma unroll
      for (int off = 1; off < 16; off <<= 1) rs += __shfl_xor(rs, off, 32);
      lsum[r] = lsum[r] * corr[r] + rs;
      m[r] = mnew;
    }

    // ---- rescale O, transpose P (C layout -> A layout) via per-wave LDS ----
    bf16_t* pw = sP + wid * 16 * LDP;
#pragma unroll
    for (int r = 0; r < 8; ++r) {
      pw[(rhi + r) * LDP + cn]      = (bf16_t)s0[r];
      pw[(rhi + r) * LDP + cn + 16] = (bf16_t)s1[r];
#pragma unroll
      for (int n = 0; n < 8; ++n) o[n][r] *= corr[r];
    }
    v16bf pa = load_a_frag(pw, LDP, lane);  // same-wave DS ops are in order

    // ---- O += P * V : K-dim = 32 key positions, 8 d-tiles ----
#pragma unroll
    for (int n = 0; n < 8; ++n) {
      v16bf bv = load_bt_frag(bufV + (n * 16) * LDV, LDV, lane);
      o[n] = wmma_bf16(pa, bv, o[n]);
    }
  }

  // ---- normalize and write AO[b, q, h*128 + d] ----
  bf16_t* aout = AO + ((size_t)b * SEQ + q0) * HIDDEN + h * HD;
#pragma unroll
  for (int r = 0; r < 8; ++r) {
    const float rinv = 1.0f / lsum[r];
#pragma unroll
    for (int n = 0; n < 8; ++n)
      aout[(size_t)(rhi + r) * HIDDEN + n * 16 + cn] = (bf16_t)(o[n][r] * rinv);
  }
}

// ---------------- host-side launch ----------------

extern "C" void kernel_launch(void* const* d_in, const int* in_sizes, int n_in,
                              void* d_out, int out_size, void* d_ws, size_t ws_size,
                              hipStream_t stream) {
  (void)in_sizes; (void)n_in; (void)out_size; (void)ws_size;

  const float* hs  = (const float*)d_in[0];   // [B, L, 2048]
  const int*   pid = (const int*)d_in[1];     // [L]
  const float* Wq  = (const float*)d_in[2];   // [2048, 2048]
  const float* Wk  = (const float*)d_in[3];   // [2048, 128]
  const float* Wv  = (const float*)d_in[4];   // [2048, 128]
  const float* Wo  = (const float*)d_in[5];   // [2048, 2048]
  float* out = (float*)d_out;                 // [B, L, 2048]

  // workspace carve-up (all sizes multiple of 256B)
  char* w = (char*)d_ws;
  bf16_t* WqT   = (bf16_t*)w;  w += (size_t)HIDDEN * (NH * HD) * 2;     // [2048 x 2048]
  bf16_t* WkvT  = (bf16_t*)w;  w += (size_t)256 * HIDDEN * 2;          // [256 x 2048]
  bf16_t* WoT   = (bf16_t*)w;  w += (size_t)HIDDEN * HIDDEN * 2;       // [2048 x 2048]
  float*  Qraw  = (float*)w;   w += (size_t)ROWS * HIDDEN * 4;         // [4096 x 2048]
  float*  KVraw = (float*)w;   w += (size_t)ROWS * 256 * 4;            // [4096 x 256]
  bf16_t* Qb    = (bf16_t*)w;  w += (size_t)BATCH * NH * SEQ * HD * 2; // [B,H,L,128]
  bf16_t* Kb    = (bf16_t*)w;  w += (size_t)BATCH * SEQ * HD * 2;      // [B,L,128]
  bf16_t* Vt    = (bf16_t*)w;  w += (size_t)BATCH * HD * SEQ * 2;      // [B,128,L]
  bf16_t* hsb   = (bf16_t*)w;  w += (size_t)ROWS * HIDDEN * 2;         // [4096 x 2048]
  bf16_t* AO    = hsb;  // hs(bf16) is dead after the QKV GEMMs; reuse for attn out

  // 1) convert hidden states to bf16
  {
    int n = ROWS * HIDDEN;
    cvt_f32_bf16_kernel<<<(n + 255) / 256, 256, 0, stream>>>(hs, hsb, n);
  }
  // 2) transpose-convert weights to bf16 (N x K)
  tcvt_f32_bf16_kernel<<<dim3(64, 64), 256, 0, stream>>>(Wq, WqT, HIDDEN, NH * HD);
  tcvt_f32_bf16_kernel<<<dim3(4, 64),  256, 0, stream>>>(Wk, WkvT, HIDDEN, HD);
  tcvt_f32_bf16_kernel<<<dim3(4, 64),  256, 0, stream>>>(Wv, WkvT + (size_t)HD * HIDDEN,
                                                         HIDDEN, HD);
  tcvt_f32_bf16_kernel<<<dim3(64, 64), 256, 0, stream>>>(Wo, WoT, NH * HD, HIDDEN);

  // 3) QKV projections (WMMA GEMM)
  gemm_bf16_kernel<<<dim3(16, 32), 256, 0, stream>>>(hsb, WqT, Qraw,
                                                     ROWS, NH * HD, HIDDEN);
  gemm_bf16_kernel<<<dim3(2, 32), 256, 0, stream>>>(hsb, WkvT, KVraw,
                                                    ROWS, 256, HIDDEN);

  // 4) RoPE + pack into attention layouts (bf16)
  rope_q_kernel<<<(BATCH * SEQ * NH * HD + 255) / 256, 256, 0, stream>>>(Qraw, pid, Qb);
  rope_kv_kernel<<<(BATCH * SEQ * 256 + 255) / 256, 256, 0, stream>>>(KVraw, pid, Kb, Vt);

  // 5) flash attention (note: AO aliases hsb, which is no longer needed)
  flash_attn_kernel<<<dim3(SEQ / 64, NH, BATCH), 128, 0, stream>>>(Qb, Kb, Vt, AO);

  // 6) output projection (WMMA GEMM) -> f32 result
  gemm_bf16_kernel<<<dim3(16, 32), 256, 0, stream>>>(AO, WoT, out,
                                                     ROWS, HIDDEN, HIDDEN);
}
